// VLMSpatialHead_76879914599138
// MI455X (gfx1250) — matrix operand order repaired and checked by reading, not compiled, not run on hardware
//
#include <hip/hip_runtime.h>
#include <hip/hip_bf16.h>
#include <math.h>

// ---------------------------------------------------------------------------
// VLM spatial head for MI455X (gfx1250, wave32, WMMA).
//   out[32,256] = segment_sum( mish(X@W1^T + b1) @ W2^T + b2 )
// X: [32768, 2048] f32.  ~155 GFLOP vs 256MB dominant read -> compute bound,
// so both GEMMs run on v_wmma_f32_16x16x32_bf16 with f32 accumulation.
// HBM floor ~11us @ 23.3 TB/s; bf16 WMMA keeps math within a few x of that.
// Double-buffered LDS staging: 1 barrier per K-step, global loads for the
// next tile overlap the current WMMAs.
// Workspace layout (needs ~69 MB):
//   [0, 4MB)        W1 in bf16        (1024 x 2048)
//   [4MB, 4.5MB)    W2 in bf16        (256  x 1024)
//   [4.5MB, 68.5MB) h  in bf16        (32768 x 1024)
// ---------------------------------------------------------------------------

typedef __bf16 bf16_t;
typedef __attribute__((ext_vector_type(16))) __bf16 v16bf;
typedef __attribute__((ext_vector_type(8)))  __bf16 v8bf;
typedef __attribute__((ext_vector_type(8)))  float  v8f;

#define H_DIM  2048
#define DMID   1024
#define DOUT   256
#define NTOK   32768
#define NIMG   32

union Frag { v16bf v; v8bf h[2]; };

__device__ __forceinline__ v16bf make_frag(v8bf lo, v8bf hi) {
  Frag f; f.h[0] = lo; f.h[1] = hi; return f.v;
}

// mish(x) = x * tanh(softplus(x)).  With t = e^x:
//   tanh(ln(1+t)) = ((1+t)^2 - 1)/((1+t)^2 + 1) = u/(u+2),  u = t*(t+2)
// One v_exp_f32 + one v_rcp_f32 + 4 VALU ops; clamp keeps u finite
// (x >= 30 -> u/(u+2) == 1.0f exactly in f32; x << 0 -> u -> 0 -> y -> 0).
__device__ __forceinline__ float mishf(float x) {
  const float t = __expf(fminf(x, 30.0f));
  const float u = t * (t + 2.0f);
  return x * u * __builtin_amdgcn_rcpf(u + 2.0f);
}

// ---------------------------------------------------------------------------
// Kernel 0: convert W1/W2 to bf16, zero pooled output.
// ---------------------------------------------------------------------------
__global__ __launch_bounds__(256) void prep_kernel(
    const float* __restrict__ W1, const float* __restrict__ W2,
    bf16_t* __restrict__ W1b, bf16_t* __restrict__ W2b,
    float* __restrict__ out)
{
  const int n1 = DMID * H_DIM;    // 2,097,152
  const int n2 = DOUT * DMID;     //   262,144
  const int no = NIMG * DOUT;     //     8,192
  int i = blockIdx.x * blockDim.x + threadIdx.x;
  if (i < n1) { W1b[i] = (bf16_t)W1[i]; return; }
  i -= n1;
  if (i < n2) { W2b[i] = (bf16_t)W2[i]; return; }
  i -= n2;
  if (i < no) out[i] = 0.0f;
}

// ---------------------------------------------------------------------------
// Kernel 1: h = mish(X @ W1^T + b1), h stored bf16.
// Block tile 128x128, K-step 32, double-buffered LDS.
// 8 waves, each wave = 32x64 (2x4 wmma tiles).
// ---------------------------------------------------------------------------
__global__ __launch_bounds__(256) void gemm1_mish_kernel(
    const float* __restrict__ X, const bf16_t* __restrict__ W1b,
    const float* __restrict__ b1, const int* __restrict__ flat_idx,
    bf16_t* __restrict__ Hout)
{
  __shared__ bf16_t As[2][128][40];   // 2 x 10 KB
  __shared__ bf16_t Bs[2][128][40];   // 2 x 10 KB

  const int bm = blockIdx.x >> 3;          // 0..255  (token tiles)
  const int bn = blockIdx.x & 7;           // 0..7    (mid tiles)
  const int m0 = bm * 128;
  const int n0 = bn * 128;

  const int tid  = threadIdx.x;
  const int lane = tid & 31;
  const int wid  = tid >> 5;               // 0..7
  const int wm   = wid & 3;                // 4 row-groups of 32
  const int wn   = wid >> 2;               // 2 col-groups of 64

  // staging role: 2 threads per row, 16 contiguous K each
  const int srow  = tid >> 1;              // 0..127
  const int halfk = (tid & 1) * 16;        // 0 or 16

  const int   grow = flat_idx[m0 + srow];  // ragged gather (identity here)
  const float*  xrow = X   + (size_t)grow * H_DIM;
  const bf16_t* wrow = W1b + (size_t)(n0 + srow) * H_DIM;

  // stage X slice (f32 -> bf16) and W1 slice (bf16 copy) into buffer `buf`
  auto stage = [&](int buf, int k0) {
    const float4* xs = (const float4*)(xrow + k0 + halfk);
    float4 f0 = xs[0], f1 = xs[1], f2 = xs[2], f3 = xs[3];
    __builtin_prefetch(xrow + k0 + 64, 0, 0);     // -> global_prefetch_b8
    bf16_t* ad = &As[buf][srow][halfk];
    ad[0]=(bf16_t)f0.x;  ad[1]=(bf16_t)f0.y;  ad[2]=(bf16_t)f0.z;  ad[3]=(bf16_t)f0.w;
    ad[4]=(bf16_t)f1.x;  ad[5]=(bf16_t)f1.y;  ad[6]=(bf16_t)f1.z;  ad[7]=(bf16_t)f1.w;
    ad[8]=(bf16_t)f2.x;  ad[9]=(bf16_t)f2.y;  ad[10]=(bf16_t)f2.z; ad[11]=(bf16_t)f2.w;
    ad[12]=(bf16_t)f3.x; ad[13]=(bf16_t)f3.y; ad[14]=(bf16_t)f3.z; ad[15]=(bf16_t)f3.w;
    const v8bf* ws = (const v8bf*)(wrow + k0 + halfk);
    *(v8bf*)&Bs[buf][srow][halfk]     = ws[0];
    *(v8bf*)&Bs[buf][srow][halfk + 8] = ws[1];
  };

  const v8f zacc = {0.f,0.f,0.f,0.f,0.f,0.f,0.f,0.f};
  v8f acc[2][4];
#pragma unroll
  for (int i = 0; i < 2; ++i)
#pragma unroll
    for (int j = 0; j < 4; ++j) acc[i][j] = zacc;

  const int a_kb = 8  * (lane >> 4);       // A-frag: two 8-elem runs
  const int b_kb = 8  * (lane >> 4);       // B-frag: two 8-elem runs
  const int frow = lane & 15;

  stage(0, 0);

  const int nk = H_DIM / 32;               // 64 K-steps
  for (int ki = 0; ki < nk; ++ki) {
    const int buf = ki & 1;
    __syncthreads();                       // staged tile `buf` now visible
    if (ki + 1 < nk) stage(buf ^ 1, (ki + 1) * 32);   // overlaps the WMMAs

    v16bf a[2], b[4];
#pragma unroll
    for (int tm = 0; tm < 2; ++tm) {
      const int r = wm * 32 + tm * 16 + frow;
      a[tm] = make_frag(*(const v8bf*)&As[buf][r][a_kb],
                        *(const v8bf*)&As[buf][r][a_kb + 16]);
    }
#pragma unroll
    for (int tn = 0; tn < 4; ++tn) {
      const int r = wn * 64 + tn * 16 + frow;
      b[tn] = make_frag(*(const v8bf*)&Bs[buf][r][b_kb],
                        *(const v8bf*)&Bs[buf][r][b_kb + 16]);
    }
#pragma unroll
    for (int tm = 0; tm < 2; ++tm)
#pragma unroll
      for (int tn = 0; tn < 4; ++tn)
        acc[tm][tn] = __builtin_amdgcn_wmma_f32_16x16x32_bf16(
            false, a[tm], false, b[tn], (short)0, acc[tm][tn], false, false);
  }

  // ---- epilogue: bias + mish -> bf16 h ----
#pragma unroll
  for (int tn = 0; tn < 4; ++tn) {
    const int col  = n0 + wn * 64 + tn * 16 + frow;
    const float bias = b1[col];
#pragma unroll
    for (int tm = 0; tm < 2; ++tm) {
      const int rbase = m0 + wm * 32 + tm * 16 + 8 * (lane >> 4);
#pragma unroll
      for (int r = 0; r < 8; ++r) {
        const float y = mishf(acc[tm][tn][r] + bias);
        Hout[(size_t)(rbase + r) * DMID + col] = (bf16_t)y;
      }
    }
  }
}

// ---------------------------------------------------------------------------
// Kernel 2: s = h @ W2^T + b2, segment-sum pool into out[32][256].
// Block tile 128 tokens x 256 out, K=1024, double-buffered LDS.
// Wave = 32x128 (2x8 wmma tiles).  Epilogue lands in a 128x256 f32 LDS tile
// (CDNA5 320KB WGP LDS), then a per-column run-length reduction over sorted
// segment_ids -> ~1-2 global atomics per column per block.
// ---------------------------------------------------------------------------
__global__ __launch_bounds__(256) void gemm2_pool_kernel(
    const bf16_t* __restrict__ Hin, const bf16_t* __restrict__ W2b,
    const float* __restrict__ b2, const int* __restrict__ seg,
    float* __restrict__ out)
{
  __shared__ bf16_t As[2][128][40];       // 20 KB
  __shared__ bf16_t Bs[2][256][40];       // 40 KB
  __shared__ float  Ob[128][DOUT + 4];    // 130 KB (stride 260 avoids conflicts)
  __shared__ int    segl[128];

  const int m0   = blockIdx.x * 128;
  const int tid  = threadIdx.x;
  const int lane = tid & 31;
  const int wid  = tid >> 5;
  const int wm   = wid & 3;               // 4 row-groups of 32
  const int wn   = wid >> 2;              // 2 col-groups of 128

  const int srow  = tid >> 1;
  const int halfk = (tid & 1) * 16;

  const bf16_t* arow = Hin + (size_t)(m0 + srow) * DMID;
  const bf16_t* brow = W2b + (size_t)tid * DMID;   // tid == W2 row 0..255

  auto stage = [&](int buf, int k0) {
    const v8bf* as = (const v8bf*)(arow + k0 + halfk);
    __builtin_prefetch(arow + k0 + 64, 0, 0);      // -> global_prefetch_b8
    *(v8bf*)&As[buf][srow][halfk]     = as[0];
    *(v8bf*)&As[buf][srow][halfk + 8] = as[1];
    const v8bf* bs = (const v8bf*)(brow + k0);
    *(v8bf*)&Bs[buf][tid][0]  = bs[0];
    *(v8bf*)&Bs[buf][tid][8]  = bs[1];
    *(v8bf*)&Bs[buf][tid][16] = bs[2];
    *(v8bf*)&Bs[buf][tid][24] = bs[3];
  };

  const v8f zacc = {0.f,0.f,0.f,0.f,0.f,0.f,0.f,0.f};
  v8f acc[2][8];
#pragma unroll
  for (int i = 0; i < 2; ++i)
#pragma unroll
    for (int j = 0; j < 8; ++j) acc[i][j] = zacc;

  const int a_kb = 8  * (lane >> 4);
  const int b_kb = 8  * (lane >> 4);
  const int frow = lane & 15;

  stage(0, 0);

  const int nk = DMID / 32;               // 32 K-steps
  for (int ki = 0; ki < nk; ++ki) {
    const int buf = ki & 1;
    __syncthreads();
    if (ki + 1 < nk) stage(buf ^ 1, (ki + 1) * 32);

    v16bf a[2], b[8];
#pragma unroll
    for (int tm = 0; tm < 2; ++tm) {
      const int r = wm * 32 + tm * 16 + frow;
      a[tm] = make_frag(*(const v8bf*)&As[buf][r][a_kb],
                        *(const v8bf*)&As[buf][r][a_kb + 16]);
    }
#pragma unroll
    for (int tn = 0; tn < 8; ++tn) {
      const int r = wn * 128 + tn * 16 + frow;
      b[tn] = make_frag(*(const v8bf*)&Bs[buf][r][b_kb],
                        *(const v8bf*)&Bs[buf][r][b_kb + 16]);
    }
#pragma unroll
    for (int tm = 0; tm < 2; ++tm)
#pragma unroll
      for (int tn = 0; tn < 8; ++tn)
        acc[tm][tn] = __builtin_amdgcn_wmma_f32_16x16x32_bf16(
            false, a[tm], false, b[tn], (short)0, acc[tm][tn], false, false);
  }

  // ---- epilogue: bias, spill block tile to LDS ----
#pragma unroll
  for (int tn = 0; tn < 8; ++tn) {
    const int col  = wn * 128 + tn * 16 + frow;
    const float bias = b2[col];
#pragma unroll
    for (int tm = 0; tm < 2; ++tm) {
      const int rbase = wm * 32 + tm * 16 + 8 * (lane >> 4);
#pragma unroll
      for (int r = 0; r < 8; ++r)
        Ob[rbase + r][col] = acc[tm][tn][r] + bias;
    }
  }
  if (tid < 128) segl[tid] = seg[m0 + tid];
  __syncthreads();

  // ---- per-column segment run reduction (segment_ids are sorted) ----
  // thread t owns column t; conflict-free LDS reads; ~1-2 atomics per column.
  float run = 0.0f;
  int   cur = segl[0];
  for (int r = 0; r < 128; ++r) {
    const int s = segl[r];
    if (s != cur) {
      atomicAdd(&out[(size_t)cur * DOUT + tid], run);
      run = 0.0f;
      cur = s;
    }
    run += Ob[r][tid];
  }
  atomicAdd(&out[(size_t)cur * DOUT + tid], run);
}

// ---------------------------------------------------------------------------
// Host-side launcher
// ---------------------------------------------------------------------------
extern "C" void kernel_launch(void* const* d_in, const int* in_sizes, int n_in,
                              void* d_out, int out_size, void* d_ws, size_t ws_size,
                              hipStream_t stream) {
  const float* X    = (const float*)d_in[0];   // [8,4096,2048] f32
  const float* W1   = (const float*)d_in[1];   // [1024,2048]
  const float* b1   = (const float*)d_in[2];   // [1024]
  const float* W2   = (const float*)d_in[3];   // [256,1024]
  const float* b2   = (const float*)d_in[4];   // [256]
  const int*   fidx = (const int*)d_in[5];     // [32768]
  const int*   segs = (const int*)d_in[6];     // [32768] sorted
  float*       out  = (float*)d_out;           // [32,256]

  char* ws = (char*)d_ws;
  bf16_t* W1b  = (bf16_t*)(ws);
  bf16_t* W2b  = (bf16_t*)(ws + (size_t)4 * 1024 * 1024);
  bf16_t* Hbuf = (bf16_t*)(ws + (size_t)4 * 1024 * 1024 + 512 * 1024);

  const int prep_n = DMID * H_DIM + DOUT * DMID + NIMG * DOUT;
  prep_kernel<<<(prep_n + 255) / 256, 256, 0, stream>>>(W1, W2, W1b, W2b, out);

  // GEMM1: 256 M-tiles x 8 N-tiles (consecutive blocks share the X tile -> L2 reuse)
  gemm1_mish_kernel<<<2048, 256, 0, stream>>>(X, W1b, b1, fidx, Hbuf);

  // GEMM2 + pooling: 256 M-tiles
  gemm2_pool_kernel<<<256, 256, 0, stream>>>(Hbuf, W2b, b2, segs, out);
}
